// SetokTokenizer_41034117546031
// MI455X (gfx1250) — matrix operand order repaired and checked by reading, not compiled, not run on hardware
//
#include <hip/hip_runtime.h>
#include <math.h>

#define NTOK 729
#define NP   768          // padded token count (multiple of 64 -> guard-free GEMM tiles)
#define CD   2048
#define DFFD 4096
#define QKVD 6144
#define HD   1024         // head dim (C / 2 heads)
#define LCL  64
#define OUTD 2048
#define EPSF 1e-5f
#define TK   32           // GEMM K-step (WMMA depth)

typedef __bf16 bf16;
typedef bf16  v16bf __attribute__((ext_vector_type(16)));
typedef float v8f   __attribute__((ext_vector_type(8)));
typedef int   v4i_vs __attribute__((vector_size(16)));   // matches async-LDS builtin param

#if defined(__HIP_DEVICE_COMPILE__) && __has_builtin(__builtin_amdgcn_global_load_async_to_lds_b128)
#define HAVE_ASYNC 1
#else
#define HAVE_ASYNC 0
#endif

__device__ __forceinline__ void wait_async0() {
#if HAVE_ASYNC
#if __has_builtin(__builtin_amdgcn_s_wait_asynccnt)
  __builtin_amdgcn_s_wait_asynccnt(0);
#else
  asm volatile("s_wait_asynccnt 0" ::: "memory");
#endif
#endif
}

// Stage a 64-row x 32-col f32 tile from global into LDS.
// 128 threads x 4 chunks of 16B = 8KB. Async path: 4 ASYNCcnt ops per wave.
__device__ __forceinline__ void stage_tile(const float* __restrict__ G, int ldg, int rbase,
                                           int kb, float* Ls, int tid) {
#pragma unroll
  for (int u = 0; u < 4; ++u) {
    int c   = tid * 4 + u;          // 0..511 chunks
    int r   = c >> 3;               // 8 chunks per 32-float row
    int col = (c & 7) * 4;
    const float* g = G + (size_t)(rbase + r) * ldg + kb + col;
    float*       l = Ls + r * TK + col;
#if HAVE_ASYNC
    __builtin_amdgcn_global_load_async_to_lds_b128(
        (__attribute__((address_space(1))) v4i_vs*)(g),
        (__attribute__((address_space(3))) v4i_vs*)(l), 0, 0);
#else
    float4 t = *(const float4*)g;
    *(float4*)l = t;
#endif
  }
}

// ---------------------------------------------------------------------------
// bf16 fragments from an LDS tile (CDNA5 16x16x32 layouts)
// A (16x32, MxK): lane m=l%16; half=l/16. elems 0..7 -> K = half*8 + i,
//                 elems 8..15 -> K = 16 + half*8 + i.
// B (32x16, KxN): lane n=l%16; half=l/16. elems 0..7 -> K = half*8 + i,
//                 elems 8..15 -> K = 16 + half*8 + i.
// ---------------------------------------------------------------------------
__device__ __forceinline__ v16bf frag_a(const float* __restrict__ Ls, int rbase, int lane) {
  int r    = rbase + (lane & 15);
  int half = lane >> 4;
  const float* p0 = Ls + r * TK + half * 8;
  const float* p1 = p0 + 16;
  v16bf f;
#pragma unroll
  for (int i = 0; i < 8; ++i) f[i] = (bf16)p0[i];
#pragma unroll
  for (int i = 0; i < 8; ++i) f[8 + i] = (bf16)p1[i];
  return f;
}

__device__ __forceinline__ v16bf frag_b(const float* __restrict__ Ls, int rbase, int lane) {
  int r    = rbase + (lane & 15);
  int half = lane >> 4;
  const float* p = Ls + r * TK + half * 8;
  v16bf f;
#pragma unroll
  for (int i = 0; i < 16; ++i) f[i] = (bf16)p[(i & 7) + (i >> 3) * 16];
  return f;
}

// ---------------------------------------------------------------------------
// GEMM: C[m,n] = act( sum_k A[m,k]*B[n,k] + bias[n] ) (+ residual[m,n])
// Workgroup (4 waves) computes a 64x64 block; each wave a 32x32 block
// (2x2 WMMA tiles). A/B K-slabs double-buffered in LDS via async copies.
// Requires M, Nn multiples of 64; K multiple of 32 (all shapes comply).
// ---------------------------------------------------------------------------
__global__ __launch_bounds__(128)
void gemm_bt(const float* __restrict__ A, int lda,
             const float* __restrict__ B, int ldb,
             const float* __restrict__ bias,
             const float* __restrict__ residual, int ldr,
             float* __restrict__ C, int ldc,
             int M, int Nn, int K, int act_gelu) {
  __shared__ float As[2][64 * TK];
  __shared__ float Bs[2][64 * TK];
  const int tid  = threadIdx.x;
  const int lane = tid & 31;
  const int wave = tid >> 5;
  const int mblk = blockIdx.x * 64;
  const int nblk = blockIdx.y * 64;
  const int wm   = (wave & 1) * 32;   // wave's 32x32 block within 64x64 tile
  const int wn   = (wave >> 1) * 32;

  v8f acc00 = {}, acc01 = {}, acc10 = {}, acc11 = {};
  const int nk = K / TK;
  int buf = 0;

  stage_tile(A, lda, mblk, 0, As[0], tid);
  stage_tile(B, ldb, nblk, 0, Bs[0], tid);
  wait_async0();
  __syncthreads();

  for (int i = 0; i < nk; ++i) {
    if (i + 1 < nk) {
      stage_tile(A, lda, mblk, (i + 1) * TK, As[buf ^ 1], tid);
      stage_tile(B, ldb, nblk, (i + 1) * TK, Bs[buf ^ 1], tid);
    }
    v16bf a0 = frag_a(As[buf], wm,      lane);
    v16bf a1 = frag_a(As[buf], wm + 16, lane);
    v16bf b0 = frag_b(Bs[buf], wn,      lane);
    v16bf b1 = frag_b(Bs[buf], wn + 16, lane);
    acc00 = __builtin_amdgcn_wmma_f32_16x16x32_bf16(false, a0, false, b0, (short)0, acc00, false, false);
    acc01 = __builtin_amdgcn_wmma_f32_16x16x32_bf16(false, a0, false, b1, (short)0, acc01, false, false);
    acc10 = __builtin_amdgcn_wmma_f32_16x16x32_bf16(false, a1, false, b0, (short)0, acc10, false, false);
    acc11 = __builtin_amdgcn_wmma_f32_16x16x32_bf16(false, a1, false, b1, (short)0, acc11, false, false);
    if (i + 1 < nk) {
      wait_async0();      // stage(i+1) landed (own wave's ASYNCcnt)
      __syncthreads();    // all waves staged & done reading buf
      buf ^= 1;
    }
  }

  // C/D layout: elem r -> row = base + (lane>=16 ? 8 : 0) + r, col = lane%16
  const int ncol = lane & 15;
  const int moff = (lane >> 4) * 8;
  v8f accs[2][2] = {{acc00, acc01}, {acc10, acc11}};
#pragma unroll
  for (int mt = 0; mt < 2; ++mt) {
#pragma unroll
    for (int nt = 0; nt < 2; ++nt) {
      int n  = nblk + wn + nt * 16 + ncol;
      int m0 = mblk + wm + mt * 16 + moff;
#pragma unroll
      for (int r = 0; r < 8; ++r) {
        float v = accs[mt][nt][r];
        if (bias) v += bias[n];
        if (act_gelu) v = 0.5f * v * (1.0f + erff(v * 0.70710678118654752f));
        if (residual) v += residual[(size_t)(m0 + r) * ldr + n];
        C[(size_t)(m0 + r) * ldc + n] = v;
      }
    }
  }
  (void)M; (void)Nn;
}

// ---------------------------------------------------------------------------
// Positional embedding add (PositionalEncoding2D, ch=1024): pad rows -> 0
// ---------------------------------------------------------------------------
__global__ __launch_bounds__(256)
void pos_add(const float* __restrict__ x, float* __restrict__ xo) {
  int t = blockIdx.x;
  int c = blockIdx.y * 256 + threadIdx.x;
  size_t o = (size_t)t * CD + c;
  if (t >= NTOK) { xo[o] = 0.f; return; }
  int i = t / 27, j = t % 27;
  int d = c, coord = i;
  if (d >= 1024) { d -= 1024; coord = j; }
  int k = d >> 1;
  float inv = expf(-(float)(2 * k) * (1.0f / 1024.0f) * 9.210340371976184f); // ln(10000)
  float si  = (float)coord * inv;
  float pos = (d & 1) ? cosf(si) : sinf(si);
  xo[o] = x[o] + pos;
}

// ---------------------------------------------------------------------------
// Row LayerNorm (one block per row)
// ---------------------------------------------------------------------------
__global__ __launch_bounds__(256)
void layernorm(const float* __restrict__ x, const float* __restrict__ g,
               const float* __restrict__ b, float* __restrict__ out, int Cc) {
  __shared__ float red[256];
  int row = blockIdx.x;
  const float* xr = x + (size_t)row * Cc;
  float s = 0.f, ss = 0.f;
  for (int c = threadIdx.x; c < Cc; c += 256) { float v = xr[c]; s += v; ss += v * v; }
  red[threadIdx.x] = s; __syncthreads();
  for (int o = 128; o > 0; o >>= 1) { if (threadIdx.x < o) red[threadIdx.x] += red[threadIdx.x + o]; __syncthreads(); }
  float mean = red[0] / Cc; __syncthreads();
  red[threadIdx.x] = ss; __syncthreads();
  for (int o = 128; o > 0; o >>= 1) { if (threadIdx.x < o) red[threadIdx.x] += red[threadIdx.x + o]; __syncthreads(); }
  float var = red[0] / Cc - mean * mean;
  float inv = rsqrtf(var + EPSF);
  for (int c = threadIdx.x; c < Cc; c += 256)
    out[(size_t)row * Cc + c] = (xr[c] - mean) * inv * g[c] + b[c];
}

// ---------------------------------------------------------------------------
// Masked softmax: P[row,:] = softmax(scale*S[row,:] over same-cluster cols)
// Pad rows/cols written as 0 so the following WMMA GEMM can run padded.
// ---------------------------------------------------------------------------
__global__ __launch_bounds__(256)
void softmax_mask(const float* __restrict__ S, int lds_, float* __restrict__ P, int ldp,
                  const int* __restrict__ idx, int Nreal, int Npad, float scale) {
  __shared__ float red[256];
  int row = blockIdx.x;
  float* pr = P + (size_t)row * ldp;
  if (row >= Nreal) { for (int c = threadIdx.x; c < Npad; c += 256) pr[c] = 0.f; return; }
  const float* sr = S + (size_t)row * lds_;
  int myc = idx ? idx[row] : 0;
  float mx = -3.4e38f;
  for (int c = threadIdx.x; c < Nreal; c += 256) {
    bool ok = idx ? (idx[c] == myc) : true;
    if (ok) mx = fmaxf(mx, sr[c] * scale);
  }
  red[threadIdx.x] = mx; __syncthreads();
  for (int o = 128; o > 0; o >>= 1) { if (threadIdx.x < o) red[threadIdx.x] = fmaxf(red[threadIdx.x], red[threadIdx.x + o]); __syncthreads(); }
  float rmax = red[0]; __syncthreads();
  float s = 0.f;
  for (int c = threadIdx.x; c < Nreal; c += 256) {
    bool ok = idx ? (idx[c] == myc) : true;
    if (ok) s += expf(sr[c] * scale - rmax);
  }
  red[threadIdx.x] = s; __syncthreads();
  for (int o = 128; o > 0; o >>= 1) { if (threadIdx.x < o) red[threadIdx.x] += red[threadIdx.x + o]; __syncthreads(); }
  float inv = 1.0f / red[0];
  for (int c = threadIdx.x; c < Npad; c += 256) {
    float v = 0.f;
    if (c < Nreal) {
      bool ok = idx ? (idx[c] == myc) : true;
      if (ok) v = expf(sr[c] * scale - rmax) * inv;
    }
    pr[c] = v;
  }
}

// dst[c][r] = src[r][c]
__global__ __launch_bounds__(256)
void transpose_mat(const float* __restrict__ src, int lds_, float* __restrict__ dst, int ldd,
                   int rows, int cols) {
  int c = blockIdx.x * 16 + (threadIdx.x & 15);
  int r = blockIdx.y * 16 + (threadIdx.x >> 4);
  if (r < rows && c < cols) dst[(size_t)c * ldd + r] = src[(size_t)r * lds_ + c];
}

// ---------------------------------------------------------------------------
// Clustering kernels
// ---------------------------------------------------------------------------
__global__ void init_scalar(float* p) { *p = 0.f; }

__global__ __launch_bounds__(256)
void dist_kernel(const float* __restrict__ gram, int ldg, float* __restrict__ dist, int ldd,
                 float* __restrict__ distmax) {
  __shared__ float red[256];
  int i = blockIdx.x;
  float invC = rsqrtf((float)CD);
  float gii = gram[(size_t)i * ldg + i];
  float lm = 0.f;
  for (int j = threadIdx.x; j < NTOK; j += 256) {
    float d2 = gii + gram[(size_t)j * ldg + j] - 2.f * gram[(size_t)i * ldg + j];
    float d = sqrtf(fmaxf(d2, 0.f)) * invC;
    dist[(size_t)i * ldd + j] = d;
    lm = fmaxf(lm, d);
  }
  red[threadIdx.x] = lm; __syncthreads();
  for (int o = 128; o > 0; o >>= 1) { if (threadIdx.x < o) red[threadIdx.x] = fmaxf(red[threadIdx.x], red[threadIdx.x + o]); __syncthreads(); }
  if (threadIdx.x == 0) atomicMax((unsigned int*)distmax, __float_as_uint(red[0]));
}

__global__ __launch_bounds__(256)
void knn_density(const float* __restrict__ dist, int ldd, float* __restrict__ density) {
  __shared__ float vals[NP];
  __shared__ float red[256];
  __shared__ int   redi[256];
  __shared__ float sumsq_s;
  int i = blockIdx.x;
  for (int j = threadIdx.x; j < NP; j += 256) vals[j] = (j < NTOK) ? dist[(size_t)i * ldd + j] : 3.4e38f;
  if (threadIdx.x == 0) sumsq_s = 0.f;
  __syncthreads();
  for (int it = 0; it < 64; ++it) {
    float bv = 3.4e38f; int bi = 0;
    for (int j = threadIdx.x; j < NP; j += 256) { float v = vals[j]; if (v < bv) { bv = v; bi = j; } }
    red[threadIdx.x] = bv; redi[threadIdx.x] = bi; __syncthreads();
    for (int o = 128; o > 0; o >>= 1) {
      if (threadIdx.x < o && red[threadIdx.x + o] < red[threadIdx.x]) {
        red[threadIdx.x] = red[threadIdx.x + o]; redi[threadIdx.x] = redi[threadIdx.x + o];
      }
      __syncthreads();
    }
    if (threadIdx.x == 0) { sumsq_s += red[0] * red[0]; vals[redi[0]] = 3.4e38f; }
    __syncthreads();
  }
  if (threadIdx.x == 0) {
    float mean = sumsq_s * (1.0f / 64.0f);
    unsigned h = (unsigned)i * 2654435761u; h ^= h >> 16; h *= 2246822519u; h ^= h >> 13;
    float u = (float)(h & 0xFFFFFFu) * (1.0f / 16777216.0f);  // deterministic tie-break
    density[i] = expf(-mean) + u * 1e-6f;
  }
}

__global__ __launch_bounds__(256)
void dmin_score(const float* __restrict__ dist, int ldd, const float* __restrict__ density,
                const float* __restrict__ distmax, float* __restrict__ score) {
  __shared__ float red[256];
  int i = blockIdx.x;
  float di = density[i];
  float best = 3.4e38f;
  for (int j = threadIdx.x; j < NTOK; j += 256)
    if (density[j] > di) best = fminf(best, dist[(size_t)i * ldd + j]);
  red[threadIdx.x] = best; __syncthreads();
  for (int o = 128; o > 0; o >>= 1) { if (threadIdx.x < o) red[threadIdx.x] = fminf(red[threadIdx.x], red[threadIdx.x + o]); __syncthreads(); }
  if (threadIdx.x == 0) score[i] = fminf(red[0], *distmax) * di;
}

__global__ __launch_bounds__(256)
void top64_centers(const float* __restrict__ score, int* __restrict__ centers) {
  __shared__ float vals[NP];
  __shared__ float red[256];
  __shared__ int   redi[256];
  __shared__ int   cen[64];
  for (int j = threadIdx.x; j < NP; j += 256) vals[j] = (j < NTOK) ? score[j] : -3.4e38f;
  __syncthreads();
  for (int it = 0; it < 64; ++it) {
    float bv = -3.4e38f; int bi = NP;
    for (int j = threadIdx.x; j < NP; j += 256) {
      float v = vals[j];
      if (v > bv || (v == bv && j < bi)) { bv = v; bi = j; }
    }
    red[threadIdx.x] = bv; redi[threadIdx.x] = bi; __syncthreads();
    for (int o = 128; o > 0; o >>= 1) {
      if (threadIdx.x < o) {
        if (red[threadIdx.x + o] > red[threadIdx.x] ||
            (red[threadIdx.x + o] == red[threadIdx.x] && redi[threadIdx.x + o] < redi[threadIdx.x])) {
          red[threadIdx.x] = red[threadIdx.x + o]; redi[threadIdx.x] = redi[threadIdx.x + o];
        }
      }
      __syncthreads();
    }
    if (threadIdx.x == 0) { cen[it] = redi[0]; vals[redi[0]] = -3.4e38f; }
    __syncthreads();
  }
  if (threadIdx.x == 0) {
    for (int a = 1; a < 64; ++a) {                    // sort ascending (jnp.sort)
      int key = cen[a], b = a - 1;
      while (b >= 0 && cen[b] > key) { cen[b + 1] = cen[b]; --b; }
      cen[b + 1] = key;
    }
    for (int a = 0; a < 64; ++a) centers[a] = cen[a];
  }
}

__global__ __launch_bounds__(256)
void assign_idx(const float* __restrict__ dist, int ldd, const int* __restrict__ centers,
                int* __restrict__ idx) {
  int j = blockIdx.x * 256 + threadIdx.x;
  if (j >= NTOK) return;
  float bv = 3.4e38f; int bc = 0;
  for (int c = 0; c < 64; ++c) {
    float v = dist[(size_t)centers[c] * ldd + j];
    if (v < bv) { bv = v; bc = c; }                   // first-min = jnp.argmin
  }
  for (int c = 0; c < 64; ++c) if (centers[c] == j) bc = c;
  idx[j] = bc;
}

__global__ __launch_bounds__(256)
void seg_mean(const float* __restrict__ feats, int ldf, const int* __restrict__ idx,
              float* __restrict__ group) {
  int c   = blockIdx.x;
  int col = blockIdx.y * 256 + threadIdx.x;
  float s = 0.f; int cnt = 0;
  for (int t = 0; t < NTOK; ++t)
    if (idx[t] == c) { s += feats[(size_t)t * ldf + col]; ++cnt; }
  group[(size_t)c * CD + col] = s / (float)cnt;
}

__global__ __launch_bounds__(256)
void write_idx_score(const int* __restrict__ idx, const float* __restrict__ score,
                     float* __restrict__ out) {
  int t = blockIdx.x * 256 + threadIdx.x;
  if (t >= NTOK) return;
  out[(size_t)LCL * OUTD + t]        = (float)idx[t];
  out[(size_t)LCL * OUTD + NTOK + t] = score[t];
}

// ---------------------------------------------------------------------------
// Host orchestration
// ---------------------------------------------------------------------------
struct BP {
  const float *ln1_g, *ln1_b, *qkv_w, *qkv_b, *proj_w, *proj_b;
  const float *ln2_g, *ln2_b, *fc1_w, *fc1_b, *fc2_w, *fc2_b;
};

static BP make_bp(void* const* d_in, int base) {
  BP p;
  p.ln1_g = (const float*)d_in[base + 0];  p.ln1_b = (const float*)d_in[base + 1];
  p.qkv_w = (const float*)d_in[base + 2];  p.qkv_b = (const float*)d_in[base + 3];
  p.proj_w = (const float*)d_in[base + 4]; p.proj_b = (const float*)d_in[base + 5];
  p.ln2_g = (const float*)d_in[base + 6];  p.ln2_b = (const float*)d_in[base + 7];
  p.fc1_w = (const float*)d_in[base + 8];  p.fc1_b = (const float*)d_in[base + 9];
  p.fc2_w = (const float*)d_in[base + 10]; p.fc2_b = (const float*)d_in[base + 11];
  return p;
}

extern "C" void kernel_launch(void* const* d_in, const int* in_sizes, int n_in,
                              void* d_out, int out_size, void* d_ws, size_t ws_size,
                              hipStream_t stream) {
  (void)in_sizes; (void)n_in; (void)out_size; (void)ws_size;

  const float* x_in = (const float*)d_in[0];
  BP inner = make_bp(d_in, 1);
  BP inter = make_bp(d_in, 13);
  const float* out_w = (const float*)d_in[25];
  const float* out_b = (const float*)d_in[26];
  float* out = (float*)d_out;

  // Workspace carve-up (floats)
  float* w = (float*)d_ws;
  size_t off = 0;
  auto alloc = [&](size_t n) { float* p = w + off; off += n; return p; };
  float* xbuf    = alloc((size_t)NP * CD);
  float* hbuf    = alloc((size_t)NP * CD);
  float* qkvbuf  = alloc((size_t)NP * QKVD);
  float* obuf    = alloc((size_t)NP * CD);
  float* ffbuf   = alloc((size_t)NP * DFFD);
  float* sbuf    = alloc((size_t)NP * NP);
  float* pbuf    = alloc((size_t)NP * NP);
  float* vtbuf   = alloc((size_t)HD * NP);
  float* gram    = alloc((size_t)NP * NP);
  float* dist    = alloc((size_t)NTOK * NP);
  float* density = alloc(NP);
  float* score   = alloc(NP);
  float* distmax = alloc(4);
  int*   centers = (int*)alloc(64);
  int*   idxbuf  = (int*)alloc(NP);
  float* group   = alloc((size_t)LCL * CD);

  auto gemm = [&](const float* A, int lda, const float* B, int ldb, const float* bias,
                  const float* res, int ldr, float* Cm, int ldc, int M, int Nn, int K, int gelu) {
    dim3 g(M / 64, Nn / 64);
    gemm_bt<<<g, 128, 0, stream>>>(A, lda, B, ldb, bias, res, ldr, Cm, ldc, M, Nn, K, gelu);
  };

  // Pre-norm transformer block (launches), Mp multiple of 64, Mreal real rows.
  auto run_block = [&](float* xb, int Mreal, int Mp, const BP& p, const int* maskidx) {
    for (int d = 0; d < 2; ++d) {
      layernorm<<<Mp, 256, 0, stream>>>(xb, p.ln1_g + (size_t)d * CD, p.ln1_b + (size_t)d * CD, hbuf, CD);
      gemm(hbuf, CD, p.qkv_w + (size_t)d * 3 * CD * CD, CD, p.qkv_b + (size_t)d * 3 * CD,
           nullptr, 0, qkvbuf, QKVD, Mp, QKVD, CD, 0);
      for (int h = 0; h < 2; ++h) {
        // scores = Qh @ Kh^T  (both row-major slices of qkv, stride QKVD)
        gemm(qkvbuf + h * HD, QKVD, qkvbuf + CD + h * HD, QKVD, nullptr,
             nullptr, 0, sbuf, Mp, Mp, Mp, HD, 0);
        softmax_mask<<<Mp, 256, 0, stream>>>(sbuf, Mp, pbuf, Mp, maskidx, Mreal, Mp, 0.03125f);
        // Vt[d,t] = V[t,d]
        transpose_mat<<<dim3(HD / 16, Mp / 16), 256, 0, stream>>>(
            qkvbuf + 2 * CD + h * HD, QKVD, vtbuf, Mp, Mp, HD);
        // o_h = P @ V = P @ Vt^T
        gemm(pbuf, Mp, vtbuf, Mp, nullptr, nullptr, 0, obuf + h * HD, CD, Mp, HD, Mp, 0);
      }
      gemm(obuf, CD, p.proj_w + (size_t)d * CD * CD, CD, p.proj_b + (size_t)d * CD,
           xb, CD, xb, CD, Mp, CD, CD, 0);
      layernorm<<<Mp, 256, 0, stream>>>(xb, p.ln2_g + (size_t)d * CD, p.ln2_b + (size_t)d * CD, hbuf, CD);
      gemm(hbuf, CD, p.fc1_w + (size_t)d * DFFD * CD, CD, p.fc1_b + (size_t)d * DFFD,
           nullptr, 0, ffbuf, DFFD, Mp, DFFD, CD, 1);
      gemm(ffbuf, DFFD, p.fc2_w + (size_t)d * CD * DFFD, DFFD, p.fc2_b + (size_t)d * CD,
           xb, CD, xb, CD, Mp, CD, DFFD, 0);
    }
  };

  // 1) x = x + pos_emb (pad rows zeroed)
  pos_add<<<dim3(NP, CD / 256), 256, 0, stream>>>(x_in, xbuf);

  // 2) DPC-KNN clustering on xbuf
  gemm(xbuf, CD, xbuf, CD, nullptr, nullptr, 0, gram, NP, NP, NP, CD, 0);   // Gram = X X^T
  init_scalar<<<1, 1, 0, stream>>>(distmax);
  dist_kernel<<<NTOK, 256, 0, stream>>>(gram, NP, dist, NP, distmax);
  knn_density<<<NTOK, 256, 0, stream>>>(dist, NP, density);
  dmin_score<<<NTOK, 256, 0, stream>>>(dist, NP, density, distmax, score);
  top64_centers<<<1, 256, 0, stream>>>(score, centers);
  assign_idx<<<(NTOK + 255) / 256, 256, 0, stream>>>(dist, NP, centers, idxbuf);

  // 3) inner transformer with same-cluster mask (in-place on xbuf)
  run_block(xbuf, NTOK, NP, inner, idxbuf);

  // 4) per-cluster mean
  seg_mean<<<dim3(LCL, CD / 256), 256, 0, stream>>>(xbuf, CD, idxbuf, group);

  // 5) inter transformer (no mask, M = 64)
  run_block(group, LCL, LCL, inter, nullptr);

  // 6) output projection -> d_out[0 .. 64*2048)
  gemm(group, CD, out_w, CD, out_b, nullptr, 0, out, OUTD, LCL, OUTD, CD, 0);

  // 7) idx_cluster (as float) and score
  write_idx_score<<<(NTOK + 255) / 256, 256, 0, stream>>>(idxbuf, score, out);
}
